// DecoderModel_74131135529564
// MI455X (gfx1250) — matrix operand order repaired and checked by reading, not compiled, not run on hardware
//
#include <hip/hip_runtime.h>
#include <hip/hip_bf16.h>

// ---------------- problem constants (from reference) ----------------
#define G    4096
#define NPG  32
#define NN   (G * NPG)        // 131072 nodes
#define DD   256              // input dim
#define SHD  256              // hidden dim
#define HDG  50
#define HDG2 100              // 2*HDG
#define HDGP 112              // padded to multiple of 16
#define HDN  3
#define HDN2 6
#define HDNP 16               // padded to 16
#define NB   2
#define TPB  16               // row tiles (of 16 nodes) per node-kernel block

typedef __attribute__((ext_vector_type(16))) _Float16 v16h;
typedef __attribute__((ext_vector_type(8)))  _Float16 h8;
typedef __attribute__((ext_vector_type(2)))  _Float16 h2;
typedef __attribute__((ext_vector_type(8)))  float    v8f;
typedef __attribute__((ext_vector_type(4)))  int      v4i;

// Row stride padding (halves) for LDS tiles: keeps 16B-aligned fragment loads
// and spreads rows across banks.
#define LPAD 8

// ---------------- async copy to LDS (CDNA5 path, guarded) ----------------
#if defined(__gfx1250__) && __has_builtin(__builtin_amdgcn_global_load_async_to_lds_b128)
#define HAVE_ASYNC 1
#else
#define HAVE_ASYNC 0
#endif

__device__ __forceinline__ void copy16_to_lds(const _Float16* src, _Float16* dst) {
#if HAVE_ASYNC
    typedef __attribute__((address_space(1))) v4i* gp_t;
    typedef __attribute__((address_space(3))) v4i* lp_t;
    __builtin_amdgcn_global_load_async_to_lds_b128((gp_t)src, (lp_t)dst, 0, 0);
#else
    *(h8*)dst = *(const h8*)src;
#endif
}

__device__ __forceinline__ void async_wait0() {
#if HAVE_ASYNC
#if __has_builtin(__builtin_amdgcn_s_wait_asynccnt)
    __builtin_amdgcn_s_wait_asynccnt(0);
#else
    asm volatile("s_wait_asynccnt 0" ::: "memory");
#endif
#endif
}

// ---- A fragment (16x32 f16) from an LDS tile, row-major, stride in halves ----
__device__ __forceinline__ v16h load_a(const _Float16* base, int stride, int lane, int kbase) {
    int row = lane & 15;
    int k0  = kbase + ((lane & 16) ? 8 : 0);
    const _Float16* p = base + row * stride + k0;
    h8 lo = *(const h8*)p;
    h8 hi = *(const h8*)(p + 16);
    v16h a;
#pragma unroll
    for (int i = 0; i < 8; ++i) { a[i] = lo[i]; a[i + 8] = hi[i]; }
    return a;
}

// ---- B fragment (32x16 f16) from transposed weights Wt[col][k], stride K halves ----
__device__ __forceinline__ v16h load_b(const _Float16* Wt_col0, int K, int lane, int kbase) {
    int n = lane & 15;
    int k = kbase + ((lane & 16) ? 8 : 0);
    const _Float16* p = Wt_col0 + (size_t)n * K + k;
    h8 lo = *(const h8*)p;
    h8 hi = *(const h8*)(p + 16);
    v16h b;
#pragma unroll
    for (int i = 0; i < 8; ++i) { b[i] = lo[i]; b[i + 8] = hi[i]; }
    return b;
}

#define WMMA(a, b, c) __builtin_amdgcn_wmma_f32_16x16x32_f16(false, (a), false, (b), (short)0, (c), false, false)

// =================== workspace init / weight prep ===================
__global__ void zero_ws_kernel(float* sum, float* counts) {
    size_t i = (size_t)blockIdx.x * 256 + threadIdx.x;
    if (i < (size_t)G * DD)            sum[i] = 0.f;
    else if (i < (size_t)G * DD + G)   counts[i - (size_t)G * DD] = 0.f;
}

// W: [NB][rows][cols] f32  ->  Wt: [NB][cols][rows] f16 (transpose + convert)
__global__ void prep_t_kernel(const float* __restrict__ W, _Float16* __restrict__ Wt,
                              int rows, int cols) {
    size_t i = (size_t)blockIdx.x * 256 + threadIdx.x;
    size_t tot = (size_t)NB * rows * cols;
    if (i >= tot) return;
    int b   = (int)(i / ((size_t)rows * cols));
    int rem = (int)(i % ((size_t)rows * cols));
    int r = rem / cols, s = rem % cols;
    Wt[((size_t)b * cols + s) * rows + r] = (_Float16)W[i];
}

// W: [NB][S][H] f32 -> Wt: [NB][HP][S] f16, rows h>=H zero-padded
__global__ void prep_pad_kernel(const float* __restrict__ W, _Float16* __restrict__ Wt,
                                int S, int H, int HP) {
    size_t i = (size_t)blockIdx.x * 256 + threadIdx.x;
    size_t tot = (size_t)NB * HP * S;
    if (i >= tot) return;
    int b   = (int)(i / ((size_t)HP * S));
    int rem = (int)(i % ((size_t)HP * S));
    int h = rem / S, s = rem % S;
    _Float16 v = (_Float16)0.f;
    if (h < H) v = (_Float16)W[((size_t)b * S + s) * H + h];
    Wt[i] = v;
}

// =================== segment-sum pooling ===================
__global__ __launch_bounds__(256) void pool_kernel(const float* __restrict__ x,
                                                   const int* __restrict__ batch,
                                                   float* __restrict__ sum,
                                                   float* __restrict__ counts) {
    int n0 = blockIdx.x * 32;
    int d  = threadIdx.x;
    __shared__ int uni;
    if (d == 0) {
        int u = 1, g = batch[n0];
        for (int j = 1; j < 32; ++j) u &= (batch[n0 + j] == g);
        uni = u;
    }
    __syncthreads();
    if (uni) {
        float s = 0.f;
#pragma unroll 4
        for (int j = 0; j < 32; ++j) s += x[(size_t)(n0 + j) * DD + d];
        int g = batch[n0];
        atomicAdd(&sum[(size_t)g * DD + d], s);
        if (d == 0) atomicAdd(&counts[g], 32.0f);
    } else {
        for (int j = 0; j < 32; ++j) {
            int g = batch[n0 + j];
            atomicAdd(&sum[(size_t)g * DD + d], x[(size_t)(n0 + j) * DD + d]);
            if (d == 0) atomicAdd(&counts[g], 1.0f);
        }
    }
}

// =================== graph MLP (both branches, per-row select) ===================
__global__ __launch_bounds__(256) void graph_kernel(
    const float* __restrict__ sum, const float* __restrict__ counts,
    const int* __restrict__ dsname,
    const _Float16* __restrict__ Wgst, const float* __restrict__ bgs,
    const _Float16* __restrict__ Wght, const float* __restrict__ bgh,
    float* __restrict__ g_head, float* __restrict__ g_var) {
    __shared__ _Float16 xh[16][DD + LPAD];
    __shared__ _Float16 hh[NB][16][SHD + LPAD];

    int g0   = blockIdx.x * 16;
    int tid  = threadIdx.x;
    int lane = tid & 31, wave = tid >> 5;

    for (int i = tid; i < 16 * DD; i += 256) {
        int r = i >> 8, c = i & (DD - 1);
        xh[r][c] = (_Float16)(sum[(size_t)(g0 + r) * DD + c] / counts[g0 + r]);
    }
    __syncthreads();

    for (int t = wave; t < NB * 16; t += 8) {
        int b = t >> 4, ct = t & 15;
        const _Float16* W = Wgst + ((size_t)b * SHD + ct * 16) * DD;
        v8f acc = {};
#pragma unroll
        for (int kc = 0; kc < DD / 32; ++kc) {
            v16h a  = load_a(&xh[0][0], DD + LPAD, lane, kc * 32);
            v16h bf = load_b(W, DD, lane, kc * 32);
            acc = WMMA(a, bf, acc);
        }
        int col = ct * 16 + (lane & 15);
        float bias = bgs[b * SHD + col];
        int rbase = (lane & 16) ? 8 : 0;
#pragma unroll
        for (int r = 0; r < 8; ++r) {
            float v = acc[r] + bias;
            hh[b][rbase + r][col] = (_Float16)(v > 0.f ? v : 0.f);
        }
    }
    __syncthreads();

    for (int t = wave; t < NB * 7; t += 8) {
        int b = t / 7, ct = t % 7;
        const _Float16* W = Wght + ((size_t)b * HDGP + ct * 16) * SHD;
        v8f acc = {};
#pragma unroll
        for (int kc = 0; kc < SHD / 32; ++kc) {
            v16h a  = load_a(&hh[b][0][0], SHD + LPAD, lane, kc * 32);
            v16h bf = load_b(W, SHD, lane, kc * 32);
            acc = WMMA(a, bf, acc);
        }
        int col = ct * 16 + (lane & 15);
        int rbase = (lane & 16) ? 8 : 0;
        if (col < HDG2) {
            float bias = bgh[b * HDG2 + col];
#pragma unroll
            for (int r = 0; r < 8; ++r) {
                int g = g0 + rbase + r;
                if (dsname[g] == b) {
                    float v = acc[r] + bias;
                    if (col < HDG) g_head[(size_t)g * HDG + col] = v;
                    else           g_var[(size_t)g * HDG + (col - HDG)] = v * v;
                }
            }
        }
    }
}

// =================== node MLP ===================
// 512 blocks x 256 threads. Each block processes TPB=16 row tiles (256 nodes).
// Layer-1 B fragments for BOTH branches live in VGPRs for the whole block
// (128 VGPRs/lane) -> weight traffic drops ~8x. Layer-2 weights staged in LDS
// once per block via GLOBAL_LOAD_ASYNC_TO_LDS_B128. x staging is software-
// pipelined: next tile's f32 values are in flight during the WMMA chain.
__global__ __launch_bounds__(256) void node_kernel(
    const float* __restrict__ x, const int* __restrict__ dsname,
    const int* __restrict__ batch,
    const _Float16* __restrict__ Wn1t, const float* __restrict__ bn1,
    const _Float16* __restrict__ Wn2t, const float* __restrict__ bn2,
    float* __restrict__ n_head, float* __restrict__ n_var) {
    __shared__ _Float16 xh[16][DD + LPAD];
    __shared__ _Float16 hh[16][SHD + LPAD];
    __shared__ _Float16 w2[NB][HDNP][SHD + LPAD];

    int tid  = threadIdx.x;
    int lane = tid & 31, wave = tid >> 5;
    int tile0 = blockIdx.x * TPB;

    // ---- stage layer-2 weights (both branches) into LDS, async ----
    {
        const int CH = NB * HDNP * (SHD / 8);       // 16B chunks = 1024
#pragma unroll
        for (int j = 0; j < CH / 256; ++j) {
            int c   = j * 256 + tid;
            int b   = c / (HDNP * (SHD / 8));
            int rem = c % (HDNP * (SHD / 8));
            int row = rem / (SHD / 8);
            int k8  = rem % (SHD / 8);
            copy16_to_lds(Wn2t + ((size_t)(b * HDNP + row) * SHD) + k8 * 8,
                          &w2[b][row][k8 * 8]);
        }
        async_wait0();
    }

    // ---- layer-1 B fragments for both branches, resident in VGPRs ----
    const _Float16* W1b0 = Wn1t + ((size_t)0 * SHD + wave * 16) * DD;
    const _Float16* W1b1 = Wn1t + ((size_t)1 * SHD + wave * 16) * DD;
    v16h B0[DD / 32], B1[DD / 32];
#pragma unroll
    for (int kc = 0; kc < DD / 32; ++kc) {
        B0[kc] = load_b(W1b0, DD, lane, kc * 32);
        B1[kc] = load_b(W1b1, DD, lane, kc * 32);
    }

    // ---- software-pipelined x staging: thread tid owns DWORD-column pairs ----
    float xr0[8], xr1[8];
    auto ldx = [&](int row0) {
#pragma unroll
        for (int j = 0; j < 8; ++j) {
            int idx = j * 256 + tid;                 // 16 rows x 128 pairs
            int r = idx >> 7, cp = idx & 127;
            const float* p = x + (size_t)(row0 + r) * DD + 2 * cp;
            xr0[j] = p[0]; xr1[j] = p[1];
        }
    };
    auto stx = [&]() {
#pragma unroll
        for (int j = 0; j < 8; ++j) {
            int idx = j * 256 + tid;
            int r = idx >> 7, cp = idx & 127;
            h2 v; v[0] = (_Float16)xr0[j]; v[1] = (_Float16)xr1[j];
            *(h2*)&xh[r][2 * cp] = v;                // one b32 LDS write/lane
        }
    };

    ldx(tile0 * 16);
    for (int t = 0; t < TPB; ++t) {
        int row0 = (tile0 + t) * 16;
        stx();
        __syncthreads();                             // xh staged; w2 visible (t==0)
        if (t + 1 < TPB) ldx((tile0 + t + 1) * 16);  // overlap with WMMA chain

        int b = dsname[batch[row0]];                 // uniform in tile (NPG=32)

        // layer 1: wave w -> hidden columns [16w,16w+16), K=256
        v8f acc = {};
        if (b == 0) {
#pragma unroll
            for (int kc = 0; kc < DD / 32; ++kc)
                acc = WMMA(load_a(&xh[0][0], DD + LPAD, lane, kc * 32), B0[kc], acc);
        } else {
#pragma unroll
            for (int kc = 0; kc < DD / 32; ++kc)
                acc = WMMA(load_a(&xh[0][0], DD + LPAD, lane, kc * 32), B1[kc], acc);
        }
        int col  = wave * 16 + (lane & 15);
        float bias = bn1[b * SHD + col];
        int rbase = (lane & 16) ? 8 : 0;
#pragma unroll
        for (int r = 0; r < 8; ++r) {
            float v = acc[r] + bias;
            hh[rbase + r][col] = (_Float16)(v > 0.f ? v : 0.f);
        }
        __syncthreads();                             // hh complete; xh free to restage

        // layer 2 (16x16 out, 6 valid cols): wave 0 only, B from LDS
        if (wave == 0) {
            v8f acc2 = {};
#pragma unroll
            for (int kc = 0; kc < SHD / 32; ++kc) {
                v16h a  = load_a(&hh[0][0], SHD + LPAD, lane, kc * 32);
                v16h bf = load_b(&w2[b][0][0], SHD + LPAD, lane, kc * 32);
                acc2 = WMMA(a, bf, acc2);
            }
            int h = lane & 15;
            if (h < HDN2) {
                float bias2 = bn2[b * HDN2 + h];
#pragma unroll
                for (int r = 0; r < 8; ++r) {
                    int node = row0 + rbase + r;
                    float v = acc2[r] + bias2;
                    if (h < HDN) n_head[(size_t)node * HDN + h] = v;
                    else         n_var[(size_t)node * HDN + (h - HDN)] = v * v;
                }
            }
        }
    }
}

// =================== host-side launcher ===================
extern "C" void kernel_launch(void* const* d_in, const int* in_sizes, int n_in,
                              void* d_out, int out_size, void* d_ws, size_t ws_size,
                              hipStream_t stream) {
    (void)in_sizes; (void)n_in; (void)out_size; (void)ws_size;
    const float* x      = (const float*)d_in[0];
    const int*   dsname = (const int*)  d_in[1];
    const int*   batch  = (const int*)  d_in[2];
    const float* Wgs    = (const float*)d_in[3];
    const float* bgs    = (const float*)d_in[4];
    const float* Wgh    = (const float*)d_in[5];
    const float* bgh    = (const float*)d_in[6];
    const float* Wn1    = (const float*)d_in[7];
    const float* bn1    = (const float*)d_in[8];
    const float* Wn2    = (const float*)d_in[9];
    const float* bn2    = (const float*)d_in[10];

    // output layout: g_head | n_head | g_var | n_var
    float* out    = (float*)d_out;
    float* g_head = out;
    float* n_head = out + (size_t)G * HDG;
    float* g_var  = n_head + (size_t)NN * HDN;
    float* n_var  = g_var + (size_t)G * HDG;

    // workspace carve-up (all offsets 32B-aligned)
    float*    sum    = (float*)d_ws;                         // G*DD f32
    float*    counts = sum + (size_t)G * DD;                 // G f32
    _Float16* Wn1t   = (_Float16*)(counts + G);              // NB*SHD*DD f16
    _Float16* Wgst   = Wn1t + (size_t)NB * SHD * DD;         // NB*SHD*DD f16
    _Float16* Wn2t   = Wgst + (size_t)NB * SHD * DD;         // NB*HDNP*SHD f16
    _Float16* Wght   = Wn2t + (size_t)NB * HDNP * SHD;       // NB*HDGP*SHD f16

    {
        int total = G * DD + G;
        zero_ws_kernel<<<(total + 255) / 256, 256, 0, stream>>>(sum, counts);
    }
    {
        int tot1 = NB * DD * SHD;
        prep_t_kernel<<<(tot1 + 255) / 256, 256, 0, stream>>>(Wn1, Wn1t, DD, SHD);
        prep_t_kernel<<<(tot1 + 255) / 256, 256, 0, stream>>>(Wgs, Wgst, DD, SHD);
        int tot2 = NB * HDNP * SHD;
        prep_pad_kernel<<<(tot2 + 255) / 256, 256, 0, stream>>>(Wn2, Wn2t, SHD, HDN2, HDNP);
        int tot3 = NB * HDGP * SHD;
        prep_pad_kernel<<<(tot3 + 255) / 256, 256, 0, stream>>>(Wgh, Wght, SHD, HDG2, HDGP);
    }
    pool_kernel<<<NN / 32, 256, 0, stream>>>(x, batch, sum, counts);
    graph_kernel<<<G / 16, 256, 0, stream>>>(sum, counts, dsname,
                                             Wgst, bgs, Wght, bgh, g_head, g_var);
    node_kernel<<<NN / (16 * TPB), 256, 0, stream>>>(x, dsname, batch,
                                                     Wn1t, bn1, Wn2t, bn2,
                                                     n_head, n_var);
}
